// Decoder_88493506167281
// MI455X (gfx1250) — compile-verified, matrix operand order checked by reading, not run on hardware
//
#include <hip/hip_runtime.h>
#include <hip/hip_bf16.h>
#include <math.h>

// Problem constants (from reference)
#define VOCAB 32000
#define EDIM  300
#define EPAD  320      // pad K to multiple of 32 for WMMA
#define HID   1024
#define ODIM  1024
#define SEQ   4096
#define H4    4096     // 4*HID
#define LSTM_BLOCKS 32

typedef __attribute__((ext_vector_type(16))) __bf16       v16bf;
typedef __attribute__((ext_vector_type(8)))  float        v8f;
typedef __attribute__((ext_vector_type(8)))  unsigned int v8u;

__device__ __forceinline__ unsigned short f2bf(float f) {
    union { float f; unsigned u; } v; v.f = f;
    unsigned u = v.u;
    unsigned r = (u + 0x7FFFu + ((u >> 16) & 1u)) >> 16;
    return (unsigned short)r;
}
__device__ __forceinline__ float bf2f(unsigned short h) {
    union { unsigned u; float f; } v; v.u = ((unsigned)h) << 16;
    return v.f;
}
__device__ __forceinline__ float sigm(float x) { return 1.f / (1.f + __expf(-x)); }

// ---------------------------------------------------------------------------
// Small elementwise kernels
// ---------------------------------------------------------------------------
__global__ void k_zero_f32(float* p, int n) {
    int i = blockIdx.x * blockDim.x + threadIdx.x;
    if (i < n) p[i] = 0.f;
}
__global__ void k_zero_i32(int* p, int n) {
    int i = blockIdx.x * blockDim.x + threadIdx.x;
    if (i < n) p[i] = 0;
}

__global__ void k_bias_sum(const float* __restrict__ a, const float* __restrict__ b,
                           float* __restrict__ o, int n) {
    int i = blockIdx.x * blockDim.x + threadIdx.x;
    if (i < n) o[i] = a[i] + b[i];
}

// gather embeddings -> bf16, pad cols 300 -> 320 with zeros
__global__ void k_embed(const float* __restrict__ emb, const int* __restrict__ tokens,
                        unsigned short* __restrict__ X) {
    int row = blockIdx.x;               // 0..SEQ-1  (tokens[:-1])
    int tok = tokens[row];
    for (int c = threadIdx.x; c < EPAD; c += blockDim.x)
        X[(size_t)row * EPAD + c] = (c < EDIM) ? f2bf(emb[(size_t)tok * EDIM + c])
                                               : (unsigned short)0;
}

// fp32 -> bf16 with optional column padding
__global__ void k_cvt_bf16(const float* __restrict__ src, unsigned short* __restrict__ dst,
                           int rows, int cols, int dcols) {
    size_t idx = (size_t)blockIdx.x * blockDim.x + threadIdx.x;
    size_t tot = (size_t)rows * dcols;
    if (idx >= tot) return;
    int r = (int)(idx / dcols), c = (int)(idx % dcols);
    dst[idx] = (c < cols) ? f2bf(src[(size_t)r * cols + c]) : (unsigned short)0;
}

// strict-triangular masked adjacency, both orientations, to bf16
// A1[i,j] = D[j,i] * (j<i)   (heads_gcn = A1 @ head_scores)
// A2[i,j] = D[i,j] * (i<j)   (mods_gcn  = A2 @ mod_scores)
__global__ void k_maskD(const float* __restrict__ dep,
                        unsigned short* __restrict__ A1, unsigned short* __restrict__ A2) {
    size_t idx = (size_t)blockIdx.x * blockDim.x + threadIdx.x;
    if (idx >= (size_t)SEQ * SEQ) return;
    int i = (int)(idx >> 12), j = (int)(idx & (SEQ - 1));
    float dji = dep[(size_t)j * (SEQ + 1) + i];
    float dij = dep[(size_t)i * (SEQ + 1) + j];
    A1[idx] = f2bf((j < i) ? dji : 0.f);
    A2[idx] = f2bf((i < j) ? dij : 0.f);
}

__global__ void k_tanh_cvt(const float* __restrict__ x, unsigned short* __restrict__ o, size_t n) {
    size_t i = (size_t)blockIdx.x * blockDim.x + threadIdx.x;
    if (i < n) o[i] = f2bf(tanhf(x[i]));
}

// ---------------------------------------------------------------------------
// bf16 fragment load (row-major source).  CDNA5 ISA 7.12.2 layout: lane half
// selects K-group; uint r holds packed K-pair at uint index
//   r + (r>=4?4:0) + (lane>=16?4:0)
// ---------------------------------------------------------------------------
__device__ __forceinline__ v16bf load_frag_rowmajor(const unsigned short* base, int hi) {
    const unsigned* p = (const unsigned*)base;
    v8u u;
#pragma unroll
    for (int r = 0; r < 8; ++r) {
        int idx = r + (r >= 4 ? 4 : 0) + hi * 4;
        u[r] = p[idx];
    }
    return __builtin_bit_cast(v16bf, u);
}
#define WMMA_BF16(A, B, C) \
    __builtin_amdgcn_wmma_f32_16x16x32_bf16(false, (A), false, (B), (short)0, (C), false, false)

// ---------------------------------------------------------------------------
// WMMA GEMM, C = A * B^T (+bias), 2x2 register-blocked: one wave computes a
// 32x32 output block (4 accumulators); 4 fragment loads feed 4 WMMAs per
// K-step.  A: MxK bf16 row-major, B: NxK bf16 row-major.  M,N mult of 32,
// K mult of 32.  Optional fp32 out (w/ accumulate), bf16 out, transposed
// bf16 out.
// ---------------------------------------------------------------------------
__global__ void k_gemm_nt(const unsigned short* __restrict__ A, int lda,
                          const unsigned short* __restrict__ B, int ldb,
                          float* __restrict__ Cf,
                          unsigned short* __restrict__ Cbf,
                          unsigned short* __restrict__ CbfT,
                          int ldc, int ldt,
                          const float* __restrict__ bias,
                          int M, int N, int K, int accum) {
    int wave = threadIdx.x >> 5, lane = threadIdx.x & 31;
    int hi = lane >> 4, l15 = lane & 15;
    int tn = N >> 5, tm = M >> 5;
    int gid = blockIdx.x * 8 + wave;
    if (gid >= tm * tn) return;                 // wave-uniform guard
    int m0 = (gid / tn) << 5, n0 = (gid % tn) << 5;
    const unsigned short* Ar0 = A + (size_t)(m0 + l15) * lda;
    const unsigned short* Ar1 = A + (size_t)(m0 + 16 + l15) * lda;
    const unsigned short* Br0 = B + (size_t)(n0 + l15) * ldb;
    const unsigned short* Br1 = B + (size_t)(n0 + 16 + l15) * ldb;
    v8f acc[2][2] = {};
    for (int k0 = 0; k0 < K; k0 += 32) {
        if (k0 + 64 < K) {                      // gfx1250 global_prefetch_b8 (near caches)
            __builtin_prefetch(Ar0 + k0 + 64, 0, 3);
            __builtin_prefetch(Br0 + k0 + 64, 0, 3);
        }
        v16bf a0 = load_frag_rowmajor(Ar0 + k0, hi);
        v16bf a1 = load_frag_rowmajor(Ar1 + k0, hi);
        v16bf b0 = load_frag_rowmajor(Br0 + k0, hi);
        v16bf b1 = load_frag_rowmajor(Br1 + k0, hi);
        acc[0][0] = WMMA_BF16(a0, b0, acc[0][0]);
        acc[0][1] = WMMA_BF16(a0, b1, acc[0][1]);
        acc[1][0] = WMMA_BF16(a1, b0, acc[1][0]);
        acc[1][1] = WMMA_BF16(a1, b1, acc[1][1]);
    }
    float bv[2];
    bv[0] = bias ? bias[n0 + l15] : 0.f;
    bv[1] = bias ? bias[n0 + 16 + l15] : 0.f;
#pragma unroll
    for (int im = 0; im < 2; ++im)
#pragma unroll
        for (int in_ = 0; in_ < 2; ++in_)
#pragma unroll
            for (int r = 0; r < 8; ++r) {
                int row = m0 + im * 16 + r + hi * 8;
                int col = n0 + in_ * 16 + l15;
                float v = acc[im][in_][r] + bv[in_];
                if (Cf) {
                    size_t o = (size_t)row * ldc + col;
                    float vv = v;
                    if (accum) vv += Cf[o];
                    Cf[o] = vv;
                }
                if (Cbf)  Cbf[(size_t)row * ldc + col] = f2bf(v);
                if (CbfT) CbfT[(size_t)col * ldt + row] = f2bf(v);
            }
}

// ---------------------------------------------------------------------------
// Persistent LSTM scan: single kernel, LSTM_BLOCKS blocks, per-step grid
// barrier (device atomics + s_sleep poll).  Each hidden unit j owns gate rows
// {j, H+j, 2H+j, 3H+j}, so the cell update is block-local; h is
// double-buffered in global by step parity and staged per-step into LDS via
// gfx1250 GLOBAL_LOAD_ASYNC_TO_LDS_B128 (ASYNCcnt + s_wait_asynccnt).
// ---------------------------------------------------------------------------
__global__ void k_lstm_all(const float* __restrict__ pregates,      // SEQ x 4H (incl. biases)
                           const unsigned short* __restrict__ Whh,  // 4H x H bf16
                           float* __restrict__ hbuf,                // 2 x H
                           float* __restrict__ c,                   // H
                           unsigned short* __restrict__ HsBf,       // SEQ x H bf16
                           int* __restrict__ cnt) {                 // SEQ, zeroed
    __shared__ float sh_h[HID];
    __shared__ float part[4][32][8];
    int tid = threadIdx.x;
    int jl = tid >> 3;              // 0..31 local hidden unit
    int ks = (tid & 7) * 128;       // K slice
    int j = blockIdx.x * 32 + jl;

    for (int t = 0; t < SEQ; ++t) {
        const float* h_in = hbuf + (size_t)(t & 1) * HID;
        float* h_out = hbuf + (size_t)((t + 1) & 1) * HID;
        const float* pg = pregates + (size_t)t * H4;

        // --- async-stage h (4KB) into LDS: wave0 issues 8 x b128 per lane ---
        if (tid < 32) {
            unsigned lds_base = (unsigned)(size_t)(&sh_h[0]) + (unsigned)tid * 16u;
#pragma unroll
            for (int q = 0; q < 8; ++q) {
                const float* g = h_in + q * 128 + tid * 4;
                unsigned la = lds_base + (unsigned)q * 512u;
                asm volatile("global_load_async_to_lds_b128 %0, %1, off"
                             :: "v"(la), "v"(g) : "memory");
            }
            asm volatile("s_wait_asynccnt 0x0" ::: "memory");
        }
        __syncthreads();

        // --- 4 gate dot-products over this thread's K slice (uint4 = 8 bf16) ---
        float s[4] = {0.f, 0.f, 0.f, 0.f};
        for (int k = ks; k < ks + 128; k += 8) {
            float hv[8];
#pragma unroll
            for (int q = 0; q < 8; ++q) hv[q] = sh_h[k + q];
#pragma unroll
            for (int g = 0; g < 4; ++g) {
                const uint4 w = *(const uint4*)(Whh + (size_t)(g * HID + j) * HID + k);
                unsigned wu[4] = {w.x, w.y, w.z, w.w};
#pragma unroll
                for (int q = 0; q < 4; ++q) {
                    union { unsigned u; float f; } lo, hi2;
                    lo.u = wu[q] << 16;
                    hi2.u = wu[q] & 0xffff0000u;
                    s[g] += lo.f * hv[2 * q] + hi2.f * hv[2 * q + 1];
                }
            }
        }
#pragma unroll
        for (int g = 0; g < 4; ++g) part[g][jl][tid & 7] = s[g];
        __syncthreads();
        if (tid < 32) {
            int j2 = blockIdx.x * 32 + tid;
            float gs[4];
#pragma unroll
            for (int g = 0; g < 4; ++g) {
                float a = pg[g * HID + j2];
#pragma unroll
                for (int q = 0; q < 8; ++q) a += part[g][tid][q];
                gs[g] = a;
            }
            float iv = sigm(gs[0]), fv = sigm(gs[1]), gv = tanhf(gs[2]), ov = sigm(gs[3]);
            float cn = fv * c[j2] + iv * gv;
            c[j2] = cn;
            float hn = ov * tanhf(cn);
            h_out[j2] = hn;
            HsBf[(size_t)t * HID + j2] = f2bf(hn);
        }
        // ---- grid barrier (release h_out, acquire for next h_in) ----
        __syncthreads();
        __threadfence();
        if (tid == 0) {
            atomicAdd(&cnt[t], 1);
            while (atomicAdd(&cnt[t], 0) < LSTM_BLOCKS)
                __builtin_amdgcn_s_sleep(1);
        }
        __syncthreads();
        __threadfence();
    }
}

// ---------------------------------------------------------------------------
// Fused output-projection + online log-sum-exp.  Block = 16 rows; 8 waves
// split the 2000 column tiles in groups of 4 (1 A-fragment feeds 4 WMMAs).
// ---------------------------------------------------------------------------
__global__ void k_loss_rows(const unsigned short* __restrict__ G,   // SEQ x HID bf16
                            const unsigned short* __restrict__ Wo,  // V x HID bf16
                            const float* __restrict__ bo,
                            float* __restrict__ row_lse) {
    __shared__ float smx[8][16], ssm[8][16];
    int wave = threadIdx.x >> 5, lane = threadIdx.x & 31;
    int hi = lane >> 4, l15 = lane & 15;
    int m0 = blockIdx.x << 4;
    const unsigned short* Ab = G + (size_t)(m0 + l15) * HID;
    float mrun[8], srun[8];
#pragma unroll
    for (int r = 0; r < 8; ++r) { mrun[r] = -3.4e38f; srun[r] = 0.f; }

    for (int ng = wave; ng < (VOCAB / 64); ng += 8) {   // 500 groups of 4 tiles
        int n0 = ng << 6;
        v8f acc[4] = {};
        for (int k0 = 0; k0 < HID; k0 += 32) {
            v16bf a = load_frag_rowmajor(Ab + k0, hi);
#pragma unroll
            for (int q = 0; q < 4; ++q) {
                v16bf b = load_frag_rowmajor(Wo + (size_t)(n0 + q * 16 + l15) * HID + k0, hi);
                acc[q] = WMMA_BF16(a, b, acc[q]);
            }
        }
#pragma unroll
        for (int q = 0; q < 4; ++q) {
            float bv = bo[n0 + q * 16 + l15];
#pragma unroll
            for (int r = 0; r < 8; ++r) {
                float x = acc[q][r] + bv;
                if (x > mrun[r]) { srun[r] = srun[r] * __expf(mrun[r] - x) + 1.f; mrun[r] = x; }
                else             { srun[r] += __expf(x - mrun[r]); }
            }
        }
    }
    // reduce across the 16 lanes of each half (rows live per lane-half)
#pragma unroll
    for (int mask = 1; mask < 16; mask <<= 1) {
#pragma unroll
        for (int r = 0; r < 8; ++r) {
            float mo = __shfl_xor(mrun[r], mask, 32);
            float so = __shfl_xor(srun[r], mask, 32);
            float mn = fmaxf(mrun[r], mo);
            srun[r] = srun[r] * __expf(mrun[r] - mn) + so * __expf(mo - mn);
            mrun[r] = mn;
        }
    }
    if (l15 == 0) {
#pragma unroll
        for (int r = 0; r < 8; ++r) { smx[wave][r + hi * 8] = mrun[r]; ssm[wave][r + hi * 8] = srun[r]; }
    }
    __syncthreads();
    if (threadIdx.x < 16) {
        float m = -3.4e38f, sacc = 0.f;
#pragma unroll
        for (int w = 0; w < 8; ++w) {
            float mw = smx[w][threadIdx.x], sw = ssm[w][threadIdx.x];
            float mn = fmaxf(m, mw);
            sacc = sacc * __expf(m - mn) + sw * __expf(mw - mn);
            m = mn;
        }
        row_lse[m0 + threadIdx.x] = m + __logf(sacc);
    }
}

// target logit: one wave per row
__global__ void k_target_logit(const unsigned short* __restrict__ G,
                               const unsigned short* __restrict__ Wo,
                               const float* __restrict__ bo,
                               const int* __restrict__ tokens,
                               float* __restrict__ row_tgt) {
    int gw = (int)((blockIdx.x * blockDim.x + threadIdx.x) >> 5);
    int lane = threadIdx.x & 31;
    if (gw >= SEQ) return;
    int t = tokens[gw + 1];
    const unsigned short* a = G + (size_t)gw * HID;
    const unsigned short* b = Wo + (size_t)t * HID;
    float s = 0.f;
    for (int k = lane; k < HID; k += 32) s += bf2f(a[k]) * bf2f(b[k]);
#pragma unroll
    for (int mask = 16; mask; mask >>= 1) s += __shfl_xor(s, mask, 32);
    if (lane == 0) row_tgt[gw] = s + bo[t];
}

__global__ void k_loss_reduce(const float* __restrict__ lse, const float* __restrict__ tgt,
                              float* __restrict__ out) {
    __shared__ float sh[256];
    float s = 0.f;
    for (int i = threadIdx.x; i < SEQ; i += 256) s += lse[i] - tgt[i];
    sh[threadIdx.x] = s;
    __syncthreads();
    for (int st = 128; st; st >>= 1) {
        if (threadIdx.x < st) sh[threadIdx.x] += sh[threadIdx.x + st];
        __syncthreads();
    }
    if (threadIdx.x == 0) out[0] = sh[0] / (float)SEQ;
}

// ---------------------------------------------------------------------------
extern "C" void kernel_launch(void* const* d_in, const int* in_sizes, int n_in,
                              void* d_out, int out_size, void* d_ws, size_t ws_size,
                              hipStream_t stream) {
    (void)in_sizes; (void)n_in; (void)out_size; (void)ws_size;
    const float* dep   = (const float*)d_in[0];
    const float* emb   = (const float*)d_in[1];
    const float* W_ih  = (const float*)d_in[2];
    const float* W_hh  = (const float*)d_in[3];
    const float* b_ih  = (const float*)d_in[4];
    const float* b_hh  = (const float*)d_in[5];
    const float* Wh    = (const float*)d_in[6];
    const float* bh    = (const float*)d_in[7];
    const float* Wm    = (const float*)d_in[8];
    const float* bm    = (const float*)d_in[9];
    const float* Wc    = (const float*)d_in[10];
    const float* bc    = (const float*)d_in[11];
    const float* Wo    = (const float*)d_in[12];
    const float* bo    = (const float*)d_in[13];
    const int*   toks  = (const int*)d_in[14];
    float* out = (float*)d_out;

    size_t off = 0;
    char* base = (char*)d_ws;
    auto alloc = [&](size_t bytes) -> void* {
        off = (off + 255) & ~(size_t)255;
        void* p = base + off;
        off += bytes;
        return p;
    };
    unsigned short* Xbf    = (unsigned short*)alloc((size_t)SEQ * EPAD * 2);
    unsigned short* Wihb   = (unsigned short*)alloc((size_t)H4 * EPAD * 2);
    float*          bsum   = (float*)alloc((size_t)H4 * 4);
    float*          preg   = (float*)alloc((size_t)SEQ * H4 * 4);
    unsigned short* Whhb   = (unsigned short*)alloc((size_t)H4 * HID * 2);
    float*          hbuf   = (float*)alloc((size_t)2 * HID * 4);
    float*          cbuf   = (float*)alloc((size_t)HID * 4);
    int*            cnt    = (int*)alloc((size_t)SEQ * 4);
    unsigned short* HsBf   = (unsigned short*)alloc((size_t)SEQ * HID * 2);
    unsigned short* Whb    = (unsigned short*)alloc((size_t)ODIM * HID * 2);
    unsigned short* Wmb    = (unsigned short*)alloc((size_t)ODIM * HID * 2);
    unsigned short* Wcb    = (unsigned short*)alloc((size_t)ODIM * HID * 2);
    unsigned short* headbT = (unsigned short*)alloc((size_t)ODIM * SEQ * 2);  // transposed
    unsigned short* modbT  = (unsigned short*)alloc((size_t)ODIM * SEQ * 2);  // transposed
    float*          gcnac  = (float*)alloc((size_t)SEQ * ODIM * 4);
    unsigned short* A1     = (unsigned short*)alloc((size_t)SEQ * SEQ * 2);
    unsigned short* A2     = (unsigned short*)alloc((size_t)SEQ * SEQ * 2);
    unsigned short* gcnb   = (unsigned short*)alloc((size_t)SEQ * ODIM * 2);
    unsigned short* Wob    = (unsigned short*)alloc((size_t)VOCAB * HID * 2);
    float*          lse    = (float*)alloc((size_t)SEQ * 4);
    float*          tgt    = (float*)alloc((size_t)SEQ * 4);

    // --- prep: biases, embeddings, weight conversion to bf16 ---
    k_bias_sum<<<(H4 + 255) / 256, 256, 0, stream>>>(b_ih, b_hh, bsum, H4);
    k_embed<<<SEQ, 128, 0, stream>>>(emb, toks, Xbf);
    {
        size_t n;
        n = (size_t)H4 * EPAD;
        k_cvt_bf16<<<(unsigned)((n + 255) / 256), 256, 0, stream>>>(W_ih, Wihb, H4, EDIM, EPAD);
        n = (size_t)H4 * HID;
        k_cvt_bf16<<<(unsigned)((n + 255) / 256), 256, 0, stream>>>(W_hh, Whhb, H4, HID, HID);
        n = (size_t)ODIM * HID;
        k_cvt_bf16<<<(unsigned)((n + 255) / 256), 256, 0, stream>>>(Wh, Whb, ODIM, HID, HID);
        k_cvt_bf16<<<(unsigned)((n + 255) / 256), 256, 0, stream>>>(Wm, Wmb, ODIM, HID, HID);
        k_cvt_bf16<<<(unsigned)((n + 255) / 256), 256, 0, stream>>>(Wc, Wcb, ODIM, HID, HID);
        n = (size_t)VOCAB * HID;
        k_cvt_bf16<<<(unsigned)((n + 255) / 256), 256, 0, stream>>>(Wo, Wob, VOCAB, HID, HID);
    }
    k_zero_f32<<<(2 * HID + 255) / 256, 256, 0, stream>>>(hbuf, 2 * HID);
    k_zero_f32<<<(HID + 255) / 256, 256, 0, stream>>>(cbuf, HID);
    k_zero_i32<<<(SEQ + 255) / 256, 256, 0, stream>>>(cnt, SEQ);

    // --- pregates = X @ W_ih^T + (b_ih+b_hh) :  4096x4096, K=320 ---
    {
        int tiles = (SEQ / 32) * (H4 / 32);
        k_gemm_nt<<<(tiles + 7) / 8, 256, 0, stream>>>(
            Xbf, EPAD, Wihb, EPAD, preg, (unsigned short*)nullptr, (unsigned short*)nullptr,
            H4, 0, bsum, SEQ, H4, EPAD, 0);
    }

    // --- persistent LSTM scan (grid barrier per step, async-LDS h staging) ---
    k_lstm_all<<<LSTM_BLOCKS, 256, 0, stream>>>(preg, Whhb, hbuf, cbuf, HsBf, cnt);

    // --- head/mod/curr score GEMMs : 4096x1024, K=1024 ---
    {
        int tiles = (SEQ / 32) * (ODIM / 32);
        // head/mod stored transposed (ODIM x SEQ) so GCN GEMMs become NT form
        k_gemm_nt<<<(tiles + 7) / 8, 256, 0, stream>>>(
            HsBf, HID, Whb, HID, (float*)nullptr, (unsigned short*)nullptr, headbT,
            ODIM, SEQ, bh, SEQ, ODIM, HID, 0);
        k_gemm_nt<<<(tiles + 7) / 8, 256, 0, stream>>>(
            HsBf, HID, Wmb, HID, (float*)nullptr, (unsigned short*)nullptr, modbT,
            ODIM, SEQ, bm, SEQ, ODIM, HID, 0);
        k_gemm_nt<<<(tiles + 7) / 8, 256, 0, stream>>>(
            HsBf, HID, Wcb, HID, gcnac, (unsigned short*)nullptr, (unsigned short*)nullptr,
            ODIM, 0, bc, SEQ, ODIM, HID, 0);
    }

    // --- masked adjacency + GCN GEMMs (NT, accumulate into curr) ---
    {
        size_t n = (size_t)SEQ * SEQ;
        k_maskD<<<(unsigned)((n + 255) / 256), 256, 0, stream>>>(dep, A1, A2);
        int tiles = (SEQ / 32) * (ODIM / 32);
        k_gemm_nt<<<(tiles + 7) / 8, 256, 0, stream>>>(
            A1, SEQ, headbT, SEQ, gcnac, (unsigned short*)nullptr, (unsigned short*)nullptr,
            ODIM, 0, (const float*)nullptr, SEQ, ODIM, SEQ, 1);
        k_gemm_nt<<<(tiles + 7) / 8, 256, 0, stream>>>(
            A2, SEQ, modbT, SEQ, gcnac, (unsigned short*)nullptr, (unsigned short*)nullptr,
            ODIM, 0, (const float*)nullptr, SEQ, ODIM, SEQ, 1);
        size_t ne = (size_t)SEQ * ODIM;
        k_tanh_cvt<<<(unsigned)((ne + 255) / 256), 256, 0, stream>>>(gcnac, gcnb, ne);
    }

    // --- fused output projection + log-sum-exp, target logits, loss ---
    k_loss_rows<<<SEQ / 16, 256, 0, stream>>>(gcnb, Wob, bo, lse);
    k_target_logit<<<(SEQ * 32) / 256, 256, 0, stream>>>(gcnb, Wob, bo, toks, tgt);
    k_loss_reduce<<<1, 256, 0, stream>>>(lse, tgt, out);
}
